// VQEmbedding_57629871177838
// MI455X (gfx1250) — matrix operand order repaired and checked by reading, not compiled, not run on hardware
//
#include <hip/hip_runtime.h>

// CDNA5 / gfx1250 WMMA + TDM types
typedef __attribute__((ext_vector_type(16))) __bf16 v16bf;
typedef __attribute__((ext_vector_type(8)))  float  v8f;
typedef unsigned int u32x4 __attribute__((ext_vector_type(4)));
typedef int          i32x4 __attribute__((ext_vector_type(4)));
typedef int          i32x8 __attribute__((ext_vector_type(8)));

#define C_DIM     256
#define K_DIM     1024
#define HW        1024
#define ROWS      64          // z rows (hw positions) per workgroup
#define ZA_STRIDE 264         // ushorts per LDS row (256 + 8 pad -> bank spread)
#define EB_STRIDE 264         // = 528 B row stride = TDM pad (128 dw interval + 4 dw pad)
#define RES_SIZE  8388608     // 32*256*32*32
#define NTHREADS  128         // 4 wave32 waves
#define NKTILES   (K_DIM / 16)

#if __has_builtin(__builtin_amdgcn_tensor_load_to_lds)
#define HAVE_TDM 1
#else
#define HAVE_TDM 0
#endif

// round-to-nearest-even fp32 -> bf16 (bit math; inputs finite)
__device__ __forceinline__ unsigned short f2bf(float f) {
    unsigned int u = __builtin_bit_cast(unsigned int, f);
    unsigned int r = u + 0x7FFFu + ((u >> 16) & 1u);
    return (unsigned short)(r >> 16);
}

union Frag { v16bf v; uint4 q[2]; };

// ---- kernel 1: emb -> bf16 copy + ||e_k||^2 ------------------------------
__global__ void __launch_bounds__(32)
vq_prep_kernel(const float* __restrict__ emb, unsigned short* __restrict__ embBf,
               float* __restrict__ enorm) {
    const int k = blockIdx.x;
    const int t = threadIdx.x;
    float s = 0.f;
    for (int c = t; c < C_DIM; c += 32) {
        float v = emb[k * C_DIM + c];
        s += v * v;
        embBf[k * C_DIM + c] = f2bf(v);
    }
    for (int m = 16; m >= 1; m >>= 1) s += __shfl_down(s, m, 32);
    if (t == 0) enorm[k] = s;
}

#if HAVE_TDM
// Issue one TDM 2-D tile load: 16 rows x 256 bf16 from embBf -> LDS,
// with 16 B LDS padding after every 512 B row (row stride 528 B).
__device__ __forceinline__ void tdm_load_tile(unsigned lds_off,
                                              const unsigned short* gsrc) {
    unsigned long long ga = (unsigned long long)gsrc;
    u32x4 g0;
    g0[0] = 1u;                                            // count=1, user mode
    g0[1] = lds_off;                                       // lds_addr (bytes)
    g0[2] = (unsigned)(ga & 0xFFFFFFFFull);                // global_addr[31:0]
    g0[3] = (unsigned)((ga >> 32) & 0x01FFFFFFull)         // global_addr[56:32]
            | 0x80000000u;                                 // type=2 ("image")
    i32x8 g1;
    g1[0] = 0x07910000;   // data_size=2B | pad_enable | pad_interval=128dw | pad_amount=4dw
    g1[1] = 0x01000000;   // tensor_dim0 = 256 (bits 79:48, low16 here)
    g1[2] = 0x04000000;   // tensor_dim1 = 1024 (bits 111:80, low16 here)
    g1[3] = 0x01000000;   // tile_dim0  = 256 (bits 127:112)
    g1[4] = 16;           // tile_dim1  = 16  (bits 143:128); tile_dim2 = 0
    g1[5] = 256;          // tensor_dim0_stride = 256 elements
    g1[6] = 0;
    g1[7] = 0;
    i32x4 gz4 = {0, 0, 0, 0};                              // groups 2/3 unused (2-D)
#if __clang_major__ >= 23
    i32x8 gz8 = {0, 0, 0, 0, 0, 0, 0, 0};
    __builtin_amdgcn_tensor_load_to_lds(g0, g1, gz4, gz4, gz8, 0);  // 6-arg toolchain
#else
    __builtin_amdgcn_tensor_load_to_lds(g0, g1, gz4, gz4, 0);       // 5-arg toolchain
#endif
}

__device__ __forceinline__ void tdm_wait0() {
#if __has_builtin(__builtin_amdgcn_s_wait_tensorcnt)
    __builtin_amdgcn_s_wait_tensorcnt(0);
#else
    asm volatile("s_wait_tensorcnt 0x0" ::: "memory");
#endif
}
#endif

// one 16x16 distance tile: 8 chained WMMAs over C=256, then min-update
__device__ __forceinline__ void wmma_tile(const unsigned short* __restrict__ arow,
                                          const unsigned short* __restrict__ brow,
                                          int aSel0, int aSel1, int bSel,
                                          float en, int kidx,
                                          float* minv, int* mini) {
    v8f acc = {0.f, 0.f, 0.f, 0.f, 0.f, 0.f, 0.f, 0.f};
#pragma unroll
    for (int cc = 0; cc < 8; ++cc) {
        Frag fa, fb;
        fa.q[0] = *(const uint4*)(arow + cc * 32 + aSel0);
        fa.q[1] = *(const uint4*)(arow + cc * 32 + aSel1);
        fb.q[0] = *(const uint4*)(brow + cc * 32 + bSel);
        fb.q[1] = *(const uint4*)(brow + cc * 32 + bSel + 8);
        acc = __builtin_amdgcn_wmma_f32_16x16x32_bf16(
            false, fa.v, false, fb.v, (short)0, acc, false, false);
    }
#pragma unroll
    for (int j = 0; j < 8; ++j) {
        float d = en - 2.0f * acc[j];        // ||z||^2 dropped (const per row)
        if (d < minv[j]) { minv[j] = d; mini[j] = kidx; }
    }
}

// ---- kernel 2: WMMA distance search + gather + scatter + loss partials --
__global__ void __launch_bounds__(NTHREADS)
vq_search_kernel(const float* __restrict__ z, const float* __restrict__ emb,
                 const unsigned short* __restrict__ embBf,
                 const float* __restrict__ enorm, float* __restrict__ out,
                 float* __restrict__ partials) {
    __shared__ __attribute__((aligned(16))) unsigned short zA[ROWS * ZA_STRIDE];
    __shared__ __attribute__((aligned(16))) unsigned short eB[2][16 * EB_STRIDE];
    __shared__ float enormS[K_DIM];
    __shared__ int   argIdx[ROWS];
    __shared__ float redBuf[NTHREADS / 32];

    const int t    = threadIdx.x;
    const int lane = t & 31;
    const int wave = t >> 5;
    const int blk  = blockIdx.x;              // 512 blocks
    const int b    = blk >> 4;                // batch index
    const int hw0  = (blk & 15) * ROWS;       // hw offset within batch
    const long long bOff = (long long)b * C_DIM * HW;

#if HAVE_TDM
    if (wave == 0)                            // prologue DMA overlaps z staging
        tdm_load_tile((unsigned)(unsigned long long)&eB[0][0], embBf);
#endif

    // Stage z tile: z_e[b][c][hw0+r] (contiguous in r) -> LDS bf16 [r][c]
    for (int i = t; i < (ROWS * C_DIM) / 4; i += NTHREADS) {
        int c  = i >> 4;            // 0..255
        int r4 = (i & 15) * 4;      // 0..60 step 4
        float4 v = *(const float4*)(z + bOff + (long long)c * HW + hw0 + r4);
        zA[(r4 + 0) * ZA_STRIDE + c] = f2bf(v.x);
        zA[(r4 + 1) * ZA_STRIDE + c] = f2bf(v.y);
        zA[(r4 + 2) * ZA_STRIDE + c] = f2bf(v.z);
        zA[(r4 + 3) * ZA_STRIDE + c] = f2bf(v.w);
    }
    for (int i = t; i < K_DIM; i += NTHREADS) enormS[i] = enorm[i];

    // Fragment addressing: contiguous k = 16*half + s for both A and B (wave32)
    const int m16  = lane & 15;
    const int half = lane >> 4;
    const unsigned short* arow = &zA[(wave * 16 + m16) * ZA_STRIDE];
    const int aSel0 = half ? 16 : 0;   // A: lanes<16 K0-15; lanes>=16 K16-31
    const int aSel1 = half ? 24 : 8;
    const int bSel  = half ? 16 : 0;   // B: lanes<16 K0-15; lanes>=16 K16-31

    float minv[8];
    int   mini[8];
#pragma unroll
    for (int j = 0; j < 8; ++j) { minv[j] = 3.4e38f; mini[j] = 0; }

#if HAVE_TDM
    // double-buffered: TDM DMA for tile kt+1 in flight while computing tile kt
    for (int kt = 0; kt < NKTILES; ++kt) {
        const int k0  = kt << 4;
        const int cur = kt & 1;
        if (wave == 0) tdm_wait0();
        __syncthreads();            // eB[cur] ready; eB[cur^1] free for next DMA
        if (wave == 0 && kt < NKTILES - 1)
            tdm_load_tile((unsigned)(unsigned long long)&eB[cur ^ 1][0],
                          embBf + (k0 + 16) * C_DIM);
        wmma_tile(arow, &eB[cur][m16 * EB_STRIDE], aSel0, aSel1, bSel,
                  enormS[k0 + m16], k0 + m16, minv, mini);
    }
#else
    // fallback: software-pipelined double-buffered coop copy
    // stage regs: 512 uint4 per tile / 128 threads = 4 per thread
    uint4 stage[4];
    {
#pragma unroll
        for (int q = 0; q < 4; ++q) {
            int i = q * NTHREADS + t;
            stage[q] = *(const uint4*)(embBf + (i >> 5) * C_DIM + (i & 31) * 8);
        }
#pragma unroll
        for (int q = 0; q < 4; ++q) {
            int i = q * NTHREADS + t;
            *(uint4*)&eB[0][(i >> 5) * EB_STRIDE + (i & 31) * 8] = stage[q];
        }
    }
    __syncthreads();
    for (int kt = 0; kt < NKTILES; ++kt) {
        const int k0  = kt << 4;
        const int cur = kt & 1;
        if (kt < NKTILES - 1) {     // issue next tile's global loads (hidden by WMMAs)
            const unsigned short* src = embBf + (k0 + 16) * C_DIM;
#pragma unroll
            for (int q = 0; q < 4; ++q) {
                int i = q * NTHREADS + t;
                stage[q] = *(const uint4*)(src + (i >> 5) * C_DIM + (i & 31) * 8);
            }
        }
        wmma_tile(arow, &eB[cur][m16 * EB_STRIDE], aSel0, aSel1, bSel,
                  enormS[k0 + m16], k0 + m16, minv, mini);
        if (kt < NKTILES - 1) {     // store staged tile into the other buffer
#pragma unroll
            for (int q = 0; q < 4; ++q) {
                int i = q * NTHREADS + t;
                *(uint4*)&eB[cur ^ 1][(i >> 5) * EB_STRIDE + (i & 31) * 8] = stage[q];
            }
        }
        __syncthreads();
    }
#endif

    // butterfly min-reduce across the 16 lanes sharing each row set
#pragma unroll
    for (int j = 0; j < 8; ++j) {
#pragma unroll
        for (int m = 8; m >= 1; m >>= 1) {
            float ov = __shfl_xor(minv[j], m, 32);
            int   oi = __shfl_xor(mini[j], m, 32);
            if (ov < minv[j] || (ov == minv[j] && oi < mini[j])) {
                minv[j] = ov; mini[j] = oi;
            }
        }
    }
    if (m16 == 0) {
#pragma unroll
        for (int j = 0; j < 8; ++j)
            argIdx[wave * 16 + half * 8 + j] = mini[j];   // rows m=j / m=j+8
    }
    __syncthreads();

    // indices output (as float; d_out is float*)
    if (t < ROWS)
        out[RES_SIZE + 1 + (long long)b * HW + hw0 + t] = (float)argIdx[t];

    // epilogue: gather emb[idx] (fp32), scatter NCHW result, accumulate loss
    const int r    = t & (ROWS - 1);
    const int coff = t >> 6;                  // 0..1
    const int myIdx = argIdx[r];
    const float* erow = emb + (long long)myIdx * C_DIM;
    float accl = 0.f;
    for (int cb = 0; cb < C_DIM; cb += 2) {
        int c = cb + coff;
        float e = erow[c];
        long long go = bOff + (long long)c * HW + hw0 + r;  // coalesced in r
        float zv = z[go];
        out[go] = e;                          // res == z_q in forward value
        float df = e - zv;
        accl += df * df;
    }
    for (int m = 16; m >= 1; m >>= 1) accl += __shfl_down(accl, m, 32);
    if (lane == 0) redBuf[wave] = accl;
    __syncthreads();
    if (t == 0) {
        float s = 0.f;
        for (int wv = 0; wv < NTHREADS / 32; ++wv) s += redBuf[wv];
        partials[blk] = s;                    // deterministic: no fp atomics
    }
}

// ---- kernel 3: reduce 512 partials -> codebook_loss ---------------------
__global__ void __launch_bounds__(256)
vq_finalize_kernel(const float* __restrict__ partials, float* __restrict__ out) {
    __shared__ float red[8];
    const int t = threadIdx.x;
    float s = partials[t] + partials[t + 256];
    for (int m = 16; m >= 1; m >>= 1) s += __shfl_down(s, m, 32);
    if ((t & 31) == 0) red[t >> 5] = s;
    __syncthreads();
    if (t == 0) {
        float tot = 0.f;
        for (int i = 0; i < 8; ++i) tot += red[i];
        // q_loss + BETA*e_loss, and fwd-value q_loss == e_loss
        out[RES_SIZE] = tot * (1.0f + 0.25f) / (float)RES_SIZE;
    }
}

extern "C" void kernel_launch(void* const* d_in, const int* in_sizes, int n_in,
                              void* d_out, int out_size, void* d_ws, size_t ws_size,
                              hipStream_t stream) {
    (void)in_sizes; (void)n_in; (void)out_size; (void)ws_size;
    const float* z   = (const float*)d_in[0];   // (32,256,32,32) fp32
    const float* emb = (const float*)d_in[1];   // (1024,256) fp32
    float* out = (float*)d_out;                 // res | loss | indices

    unsigned short* embBf = (unsigned short*)d_ws;                       // 512 KB bf16
    float* enorm    = (float*)((char*)d_ws + (size_t)K_DIM * C_DIM * 2); // 4 KB
    float* partials = enorm + K_DIM;                                     // 2 KB

    vq_prep_kernel<<<K_DIM, 32, 0, stream>>>(emb, embBf, enorm);
    vq_search_kernel<<<512, NTHREADS, 0, stream>>>(z, emb, embBf, enorm, out, partials);
    vq_finalize_kernel<<<1, 256, 0, stream>>>(partials, out);
}
